// ScaledDotProductAttention_66434554134747
// MI455X (gfx1250) — matrix operand order repaired and checked by reading, not compiled, not run on hardware
//
#include <hip/hip_runtime.h>
#include <hip/hip_bf16.h>
#include <math.h>

typedef __attribute__((ext_vector_type(16))) __bf16 v16bf;
typedef __attribute__((ext_vector_type(8)))  float  v8f;

#define B_  2
#define H_  16
#define S_  2048
#define D_  64
#define BH_ (B_ * H_)
#define PSTR 36   // padded f32 stride for per-wave P staging (mult of 4, odd*4 banks)

// scale * log2(e): softmax exp computed as exp2(score * SCALE_LOG2E)
#define SCALE_LOG2E (0.125f * 1.44269504088896340736f)

// xor-butterfly step via ds_swizzle (group-of-32 encoding: [14:10]=xor, [9:5]=or, [4:0]=and)
__device__ __forceinline__ float swz_xor_add(float v, int xmask) {
  int iv = __float_as_int(v);
  int sw;
  switch (xmask) {   // ds_swizzle offset must be an immediate
    case 1: sw = __builtin_amdgcn_ds_swizzle(iv, (1 << 10) | 0x1f); break;
    case 2: sw = __builtin_amdgcn_ds_swizzle(iv, (2 << 10) | 0x1f); break;
    case 4: sw = __builtin_amdgcn_ds_swizzle(iv, (4 << 10) | 0x1f); break;
    default: sw = __builtin_amdgcn_ds_swizzle(iv, (8 << 10) | 0x1f); break;
  }
  return v + __int_as_float(sw);
}

// ---------------------------------------------------------------------------
// Prep: V[bh][s][d] f32  ->  Vt[bh][d][s] bf16   (so PV B-fragments are
// contiguous 16-element loads per lane)
// ---------------------------------------------------------------------------
__global__ __launch_bounds__(256) void vtrans_kernel(const float* __restrict__ v,
                                                     __bf16* __restrict__ vt) {
  __shared__ float st[64 * 65];
  const int bh = blockIdx.x >> 5;        // 32 s-chunks per bh
  const int s0 = (blockIdx.x & 31) * 64;
  const int t  = threadIdx.x;

  const float* vb = v + (size_t)bh * S_ * D_ + (size_t)s0 * D_;
#pragma unroll
  for (int r = 0; r < 4; ++r) {
    int lin = r * 1024 + t * 4;          // linear over 64x64 tile, d fastest
    int i = lin >> 6, d = lin & 63;
    float4 val = *(const float4*)(vb + lin);
    st[(d + 0) * 65 + i] = val.x;
    st[(d + 1) * 65 + i] = val.y;
    st[(d + 2) * 65 + i] = val.z;
    st[(d + 3) * 65 + i] = val.w;
  }
  __syncthreads();

  const int d  = t >> 2;
  const int qq = t & 3;
  v16bf pack;
#pragma unroll
  for (int e = 0; e < 16; ++e) pack[e] = (__bf16)st[d * 65 + qq * 16 + e];
  *(v16bf*)(vt + (size_t)bh * D_ * S_ + (size_t)d * S_ + s0 + qq * 16) = pack;
}

// ---------------------------------------------------------------------------
// One 16x16 tile of exp(mask ? -inf : scale*QK^T), columns [col0, col0+16).
// Returns per-lane v8f in C-matrix layout.
// ---------------------------------------------------------------------------
__device__ __forceinline__ v8f score_exp_tile(const float* __restrict__ kbase,
                                              v16bf aq0, v16bf aq1,
                                              const unsigned char* __restrict__ smask,
                                              int col0, int lane) {
  const int n  = lane & 15;
  const int kh = lane >> 4;
  const float* kr = kbase + (size_t)(col0 + n) * D_ + kh * 16;

  float tf[16];
  v16bf b0, b1;
  // B fragment, K-dim chunk d=0..31: element e <-> K = kh*16 + e (contiguous)
  *(float4*)(tf + 0)  = *(const float4*)(kr + 0);
  *(float4*)(tf + 4)  = *(const float4*)(kr + 4);
  *(float4*)(tf + 8)  = *(const float4*)(kr + 8);
  *(float4*)(tf + 12) = *(const float4*)(kr + 12);
#pragma unroll
  for (int e = 0; e < 16; ++e) b0[e] = (__bf16)tf[e];
  // chunk d=32..63
  *(float4*)(tf + 0)  = *(const float4*)(kr + 32 + 0);
  *(float4*)(tf + 4)  = *(const float4*)(kr + 32 + 4);
  *(float4*)(tf + 8)  = *(const float4*)(kr + 32 + 8);
  *(float4*)(tf + 12) = *(const float4*)(kr + 32 + 12);
#pragma unroll
  for (int e = 0; e < 16; ++e) b1[e] = (__bf16)tf[e];

  v8f c = {};
  c = __builtin_amdgcn_wmma_f32_16x16x32_bf16(false, aq0, false, b0, (short)0, c, false, false);
  c = __builtin_amdgcn_wmma_f32_16x16x32_bf16(false, aq1, false, b1, (short)0, c, false, false);

  const int rl_base = (lane < 16) ? 0 : 8;
  const int col = col0 + n;
  v8f p;
#pragma unroll
  for (int i = 0; i < 8; ++i) {
    float e2 = __builtin_amdgcn_exp2f(c[i] * SCALE_LOG2E);  // exp(scale*s), one mul
    unsigned char mb = smask[(rl_base + i) * S_ + col];
    p[i] = mb ? 0.0f : e2;   // masked -> exp(-1e9) == 0
  }
  return p;
}

// ---------------------------------------------------------------------------
// Attention: one block = 16 query rows of one (b,h).  8 waves.
// ---------------------------------------------------------------------------
__global__ __launch_bounds__(256) void attn_kernel(const float* __restrict__ q,
                                                   const float* __restrict__ k,
                                                   const unsigned char* __restrict__ mask,
                                                   const __bf16* __restrict__ vt,
                                                   float* __restrict__ out,
                                                   float* __restrict__ atten) {
  __shared__ uint4 smask4[16 * S_ / 16];          // 16 rows x 2048 bytes = 32KB
  __shared__ float s_pst[8 * 16 * PSTR];          // per-wave P staging
  __shared__ float s_out[16 * D_];                // 4KB output accumulator
  __shared__ float s_rowsum[16];

  const int t    = threadIdx.x;
  const int lane = t & 31;
  const int w    = t >> 5;

  const int bh    = blockIdx.x >> 7;              // 128 row-blocks per bh
  const int row0  = (blockIdx.x & 127) * 16;
  const int batch = bh >> 4;                      // H = 16

  const float* qb = q + (size_t)bh * S_ * D_;
  const float* kb = k + (size_t)bh * S_ * D_;
  const __bf16* vtb = vt + (size_t)bh * D_ * S_;
  float* attb = atten + (size_t)bh * S_ * S_ + (size_t)row0 * S_;
  const unsigned char* smask = (const unsigned char*)smask4;

  // ---- stage mask rows (contiguous 32KB) via async LDS copy ----
  {
    const uint4* gm = (const uint4*)(mask + (size_t)batch * S_ * S_ + (size_t)row0 * S_);
    for (int j = t; j < 2048; j += 256) {
      unsigned long long gaddr = (unsigned long long)(uintptr_t)(gm + j);
      unsigned int ldsoff = (unsigned int)(uintptr_t)(&smask4[j]);  // low 32b = LDS offset
      asm volatile("global_load_async_to_lds_b128 %0, %1, off"
                   :: "v"(ldsoff), "v"(gaddr) : "memory");
    }
    for (int j = t; j < 16 * D_; j += 256) s_out[j] = 0.0f;
    if (t < 16) s_rowsum[t] = 0.0f;
#if __has_builtin(__builtin_amdgcn_s_wait_asynccnt)
    __builtin_amdgcn_s_wait_asynccnt(0);
#else
    asm volatile("s_wait_asynccnt 0x0" ::: "memory");
#endif
  }
  __syncthreads();

  // ---- Q A-fragments (rows row0..row0+15, K-chunks d=0..31 and 32..63) ----
  const int m  = lane & 15;
  const int kh = lane >> 4;
  v16bf aq0, aq1;
  {
    const float* qr = qb + (size_t)(row0 + m) * D_;
    float tf[16];
#pragma unroll
    for (int chunk = 0; chunk < 2; ++chunk) {
      int d0 = chunk * 32;
      *(float4*)(tf + 0)  = *(const float4*)(qr + d0 + kh * 16);       // e=0..15 -> K=kh*16+e
      *(float4*)(tf + 4)  = *(const float4*)(qr + d0 + kh * 16 + 4);
      *(float4*)(tf + 8)  = *(const float4*)(qr + d0 + kh * 16 + 8);   // (contiguous)
      *(float4*)(tf + 12) = *(const float4*)(qr + d0 + kh * 16 + 12);
      v16bf a;
#pragma unroll
      for (int e = 0; e < 16; ++e) a[e] = (__bf16)tf[e];
      if (chunk == 0) aq0 = a; else aq1 = a;
    }
  }

  // ---- Phase 1: row sums of exp(scores) over all 2048 columns ----
  v8f rsum = {};
  for (int it = 0; it < 16; ++it) {
    int col0 = (w + 8 * it) * 16;
    v8f p = score_exp_tile(kb, aq0, aq1, smask, col0, lane);
    rsum += p;
  }
#pragma unroll
  for (int i = 0; i < 8; ++i) {
    float vsum = rsum[i];
    vsum = swz_xor_add(vsum, 1);
    vsum = swz_xor_add(vsum, 2);
    vsum = swz_xor_add(vsum, 4);
    vsum = swz_xor_add(vsum, 8);   // full reduce within each 16-lane half
    if (lane == 0)  atomicAdd(&s_rowsum[i], vsum);
    if (lane == 16) atomicAdd(&s_rowsum[8 + i], vsum);
  }
  __syncthreads();

  const int rl_base = (lane < 16) ? 0 : 8;
  float rinv[8];
#pragma unroll
  for (int i = 0; i < 8; ++i) rinv[i] = 1.0f / s_rowsum[rl_base + i];

  // ---- Phase 2: recompute p, normalize, write atten, and P@V via WMMA ----
  v8f acc[4] = {v8f{}, v8f{}, v8f{}, v8f{}};
  float* pst = s_pst + w * 16 * PSTR;
  const int n = lane & 15;

  for (int it = 0; it < 8; ++it) {
    const int c0 = (w + 8 * it) * 32;   // 32-column K-chunk for PV
#pragma unroll
    for (int sub = 0; sub < 2; ++sub) {
      const int col0 = c0 + sub * 16;
      v8f p = score_exp_tile(kb, aq0, aq1, smask, col0, lane);
#pragma unroll
      for (int i = 0; i < 8; ++i) {
        float pn = p[i] * rinv[i];
        attb[(size_t)(rl_base + i) * S_ + col0 + n] = pn;       // normalized atten
        pst[(rl_base + i) * PSTR + sub * 16 + n] = pn;          // stage for A-frag
      }
    }
    __syncthreads();  // uniform trip count across waves

    // P A-fragment (16 rows x 32 cols, bf16) from staged LDS
    float tf[16];
    *(float4*)(tf + 0)  = *(const float4*)(pst + m * PSTR + kh * 16);
    *(float4*)(tf + 4)  = *(const float4*)(pst + m * PSTR + kh * 16 + 4);
    *(float4*)(tf + 8)  = *(const float4*)(pst + m * PSTR + kh * 16 + 8);
    *(float4*)(tf + 12) = *(const float4*)(pst + m * PSTR + kh * 16 + 12);
    v16bf ap;
#pragma unroll
    for (int e = 0; e < 16; ++e) ap[e] = (__bf16)tf[e];

    // V B-fragments: contiguous 16 bf16 per lane from Vt[bh][d][s]
#pragma unroll
    for (int j = 0; j < 4; ++j) {
      const v16bf bv = *(const v16bf*)(vtb + (size_t)(j * 16 + n) * S_ + c0 + kh * 16);
      acc[j] = __builtin_amdgcn_wmma_f32_16x16x32_bf16(false, ap, false, bv, (short)0, acc[j], false, false);
    }
    __syncthreads();  // protect pstage reuse next iteration
  }

  // ---- combine wave partials and write out ----
#pragma unroll
  for (int j = 0; j < 4; ++j)
#pragma unroll
    for (int i = 0; i < 8; ++i)
      atomicAdd(&s_out[(rl_base + i) * D_ + j * 16 + n], acc[j][i]);
  __syncthreads();

  float* ob = out + (size_t)bh * S_ * D_ + (size_t)row0 * D_;
  *(float4*)(ob + t * 4) = *(const float4*)(s_out + t * 4);   // 1024 floats, 256 threads
}

// ---------------------------------------------------------------------------
extern "C" void kernel_launch(void* const* d_in, const int* in_sizes, int n_in,
                              void* d_out, int out_size, void* d_ws, size_t ws_size,
                              hipStream_t stream) {
  const float* q = (const float*)d_in[0];
  const float* k = (const float*)d_in[1];
  const float* v = (const float*)d_in[2];
  const unsigned char* mask = (const unsigned char*)d_in[3];  // jnp bool -> 1 byte

  float* out   = (float*)d_out;
  float* atten = out + (size_t)B_ * H_ * S_ * D_;             // tuple order: (out, atten)
  __bf16* vt   = (__bf16*)d_ws;                               // 8 MB bf16 V-transpose

  vtrans_kernel<<<dim3(BH_ * 32), dim3(256), 0, stream>>>(v, vt);
  attn_kernel<<<dim3(BH_ * 128), dim3(256), 0, stream>>>(q, k, mask, vt, out, atten);
}
